// STGAT_77129022701698
// MI455X (gfx1250) — compile-verified, matrix operand order checked by reading, not run on hardware
//
#include <hip/hip_runtime.h>
#include <hip/hip_bf16.h>

// ---------------------------------------------------------------------------
// STGAT forward for MI455X (gfx1250, wave32).
// Dense GEMMs -> v_wmma_f32_16x16x32_bf16, 2x2 register tiling, all operand
// fragments loaded as contiguous b128 (A row-major [M,K], B stored as [N,K]).
// Attention softmax / scatter / LSTM recurrence / readout -> exact fp32 VALU.
// ---------------------------------------------------------------------------

typedef __bf16 bf16_t;
typedef __attribute__((ext_vector_type(16))) __bf16 v16bf;
typedef __attribute__((ext_vector_type(8)))  __bf16 v8bf;
typedef __attribute__((ext_vector_type(8)))  float  v8f;

#define N_NODES 16000
#define FIN     96
#define HEADS   4
#define HDIM    384          // HEADS*FIN
#define T_STEPS 12
#define BATCH   32
#define NNODE   500          // nodes per batch element
#define FEAT    8
#define SEQ_IN  4000         // NNODE*FEAT
#define TBROWS  384          // T_STEPS*BATCH
#define HH      256          // hidden size (both LSTMs)
#define G4      1024         // 4*HH
#define OUTC    1000         // NNODE*2

// ---------------------------- utility kernels ------------------------------

__global__ void fill_f32(float* p, float v, int n) {
    int i = blockIdx.x * blockDim.x + threadIdx.x;
    if (i < n) p[i] = v;
}

__global__ void fill_u32(unsigned* p, unsigned v, int n) {
    int i = blockIdx.x * blockDim.x + threadIdx.x;
    if (i < n) p[i] = v;
}

__global__ void cvt_f32_bf16(const float* __restrict__ in, bf16_t* __restrict__ out, int n) {
    int i = blockIdx.x * blockDim.x + threadIdx.x;
    if (i < n) out[i] = (bf16_t)in[i];
}

// out[c*rows + r] = (bf16) in[r*cols + c]   (forms [N,K] from a [K,N] weight)
__global__ void cvt_f32_bf16_T(const float* __restrict__ in, bf16_t* __restrict__ out,
                               int rows, int cols) {
    int i = blockIdx.x * blockDim.x + threadIdx.x;
    if (i >= rows * cols) return;
    int r = i / cols, c = i - r * cols;
    out[c * rows + r] = (bf16_t)in[i];
}

__global__ void add_bias_rows(float* __restrict__ x, const float* __restrict__ b, int n, int C) {
    int i = blockIdx.x * blockDim.x + threadIdx.x;
    if (i < n) x[i] += b[i % C];
}

// ------------------------- WMMA bf16 GEMM (wave32) -------------------------
// C[M,N] (f32) = A[M,K] (bf16 row-major) * Bt[N,K] (bf16 row-major, i.e. B^T)
// One wave per 32x32 output tile: 2x2 accumulators, 4 v_wmma per K-step.
// Fragment striping per CDNA5 ISA 7.12.2 (16-bit A 16x32):
//   lane[3:0] = m (or n), lane[4]=hi; elems 0..7 <- K run [k0+8*hi .. +7],
//   elems 8..15 <- K run [k0+16+8*hi .. +7]  -> two contiguous b128 loads.
// C/D: elem r -> row = r + 8*hi, col = lane[3:0].
__device__ __forceinline__ v16bf frag_cat(v8bf lo, v8bf hi2) {
    return __builtin_shufflevector(lo, hi2, 0, 1, 2, 3, 4, 5, 6, 7,
                                   8, 9, 10, 11, 12, 13, 14, 15);
}

__global__ void wmma_gemm_bf16_nt(const bf16_t* __restrict__ A, const bf16_t* __restrict__ Bt,
                                  float* __restrict__ C, int M, int N, int K) {
    const int lane = threadIdx.x & 31;
    const int lm   = lane & 15;
    const int hi   = lane >> 4;
    const int tn   = blockIdx.x * 32;
    const int tm   = blockIdx.y * 32;

    const v8bf* A0 = (const v8bf*)(A  + (size_t)(tm + lm)      * K);
    const v8bf* A1 = (const v8bf*)(A  + (size_t)(tm + 16 + lm) * K);
    const v8bf* B0 = (const v8bf*)(Bt + (size_t)(tn + lm)      * K);
    const v8bf* B1 = (const v8bf*)(Bt + (size_t)(tn + 16 + lm) * K);

    v8f acc00 = {}, acc01 = {}, acc10 = {}, acc11 = {};
    const int kv = K >> 3;                       // v8bf elements per row
    for (int i = 0; i < kv; i += 4) {            // 32 K-values per step
        const int i0 = i + hi;                   // elems 0..7
        const int i1 = i + 2 + hi;               // elems 8..15
        v16bf a0 = frag_cat(A0[i0], A0[i1]);
        v16bf a1 = frag_cat(A1[i0], A1[i1]);
        v16bf b0 = frag_cat(B0[i0], B0[i1]);
        v16bf b1 = frag_cat(B1[i0], B1[i1]);
        if (i + 8 <= kv) {                       // pull next K-slab toward cache
            __builtin_prefetch(&A0[i0 + 4], 0, 0);
            __builtin_prefetch(&B0[i0 + 4], 0, 0);
        }
        acc00 = __builtin_amdgcn_wmma_f32_16x16x32_bf16(false, a0, false, b0, (short)0, acc00, false, false);
        acc01 = __builtin_amdgcn_wmma_f32_16x16x32_bf16(false, a0, false, b1, (short)0, acc01, false, false);
        acc10 = __builtin_amdgcn_wmma_f32_16x16x32_bf16(false, a1, false, b0, (short)0, acc10, false, false);
        acc11 = __builtin_amdgcn_wmma_f32_16x16x32_bf16(false, a1, false, b1, (short)0, acc11, false, false);
    }

    const int n0 = tn + lm, n1 = n0 + 16;
#pragma unroll
    for (int r = 0; r < 8; ++r) {
        const int m0 = tm + r + 8 * hi;
        const int m1 = m0 + 16;
        C[(size_t)m0 * N + n0] = acc00[r];
        C[(size_t)m0 * N + n1] = acc01[r];
        C[(size_t)m1 * N + n0] = acc10[r];
        C[(size_t)m1 * N + n1] = acc11[r];
    }
}

// ------------------------------- GAT pieces --------------------------------

// alpha_src/dst[n,h] = sum_c h[n,h,c] * att[h,c]
__global__ void gat_alpha(const float* __restrict__ h,
                          const float* __restrict__ att_src,
                          const float* __restrict__ att_dst,
                          float* __restrict__ as_, float* __restrict__ ad_) {
    int i = blockIdx.x * blockDim.x + threadIdx.x;     // n*HEADS + hd
    if (i >= N_NODES * HEADS) return;
    int n = i >> 2, hd = i & 3;
    const float* hp = h + n * HDIM + hd * FIN;
    const float* s  = att_src + hd * FIN;
    const float* d  = att_dst + hd * FIN;
    float accs = 0.f, accd = 0.f;
#pragma unroll 8
    for (int c = 0; c < FIN; ++c) { accs += hp[c] * s[c]; accd += hp[c] * d[c]; }
    as_[i] = accs; ad_[i] = accd;
}

__device__ __forceinline__ unsigned f32_order(float f) {
    unsigned u = __float_as_uint(f);
    return (u & 0x80000000u) ? ~u : (u | 0x80000000u);
}
__device__ __forceinline__ float f32_unorder(unsigned u) {
    return (u & 0x80000000u) ? __uint_as_float(u & 0x7FFFFFFFu) : __uint_as_float(~u);
}

// pass 1: e = leaky_relu(as[src]+ad[dst], 0.2); segment-max into emax[dst]
__global__ void gat_edge_logits(const int* __restrict__ ei, int E,
                                const float* __restrict__ as_, const float* __restrict__ ad_,
                                float* __restrict__ eBuf, unsigned* __restrict__ emaxU) {
    int idx = blockIdx.x * blockDim.x + threadIdx.x;
    int Eall = E + N_NODES;
    if (idx >= Eall) return;
    int src = (idx < E) ? ei[idx]     : (idx - E);     // self loops appended
    int dst = (idx < E) ? ei[E + idx] : (idx - E);
#pragma unroll
    for (int hd = 0; hd < HEADS; ++hd) {
        float e = as_[src * HEADS + hd] + ad_[dst * HEADS + hd];
        e = (e > 0.f) ? e : 0.2f * e;
        eBuf[idx * HEADS + hd] = e;
        atomicMax(&emaxU[dst * HEADS + hd], f32_order(e));
    }
}

// pass 2: ee = exp(e - emax[dst]); denom[dst] += ee
__global__ void gat_edge_exp(const int* __restrict__ ei, int E,
                             float* __restrict__ eBuf, const unsigned* __restrict__ emaxU,
                             float* __restrict__ denom) {
    int idx = blockIdx.x * blockDim.x + threadIdx.x;
    int Eall = E + N_NODES;
    if (idx >= Eall) return;
    int dst = (idx < E) ? ei[E + idx] : (idx - E);
#pragma unroll
    for (int hd = 0; hd < HEADS; ++hd) {
        float m  = f32_unorder(emaxU[dst * HEADS + hd]);
        float ee = expf(eBuf[idx * HEADS + hd] - m);
        eBuf[idx * HEADS + hd] = ee;
        atomicAdd(&denom[dst * HEADS + hd], ee);
    }
}

// pass 3: out[dst,c] += (1/H) * sum_h (ee/denom) * hfeat[src,h,c]
__global__ void gat_scatter(const int* __restrict__ ei, int E,
                            const float* __restrict__ eBuf, const float* __restrict__ denom,
                            const float* __restrict__ h, float* __restrict__ out) {
    int i = blockIdx.x * blockDim.x + threadIdx.x;     // (E+N)*FIN threads
    int Eall = E + N_NODES;
    if (i >= Eall * FIN) return;
    int idx = i / FIN, c = i - idx * FIN;
    int src = (idx < E) ? ei[idx]     : (idx - E);
    int dst = (idx < E) ? ei[E + idx] : (idx - E);
    float v = 0.f;
#pragma unroll
    for (int hd = 0; hd < HEADS; ++hd) {
        float a = eBuf[idx * HEADS + hd] / (denom[dst * HEADS + hd] + 1e-16f);
        v += a * h[src * HDIM + hd * FIN + c];
    }
    atomicAdd(&out[dst * FIN + c], 0.25f * v);
}

// ------------------------------ reshape to seq -----------------------------
// Xseq[t, b, v*8+f] = x2[b*500+v, t*8+f]   (stored bf16 for the big GEMM)
__global__ void reshape_to_seq(const float* __restrict__ x2, bf16_t* __restrict__ Xb) {
    int i = blockIdx.x * blockDim.x + threadIdx.x;     // T*B*SEQ_IN
    if (i >= T_STEPS * BATCH * SEQ_IN) return;
    int t   = i / (BATCH * SEQ_IN);
    int rem = i - t * (BATCH * SEQ_IN);
    int b   = rem / SEQ_IN;
    int q   = rem - b * SEQ_IN;
    int v   = q >> 3, f = q & 7;
    Xb[i] = (bf16_t)x2[(b * NNODE + v) * FIN + t * FEAT + f];
}

// ------------------------------- LSTM step ---------------------------------
// One workgroup per batch row b (1024 thr = 32 waves on one WGP).
// gates precomputed input projection G[(t*B+b), j]; recurrent 256-dot in fp32.
__device__ __forceinline__ float sigmf(float x) { return 1.f / (1.f + expf(-x)); }

__global__ void lstm_step(const float* __restrict__ G, int t,
                          const float* __restrict__ wh,
                          const float* __restrict__ bi, const float* __restrict__ bh,
                          float* __restrict__ hstate, float* __restrict__ cstate,
                          float* __restrict__ hs_out) {
    __shared__ float hsh[HH];
    __shared__ float gates[G4];
    int b = blockIdx.x, j = threadIdx.x;
    for (int k = j; k < HH; k += blockDim.x) hsh[k] = hstate[b * HH + k];
    __syncthreads();

    float acc = G[(t * BATCH + b) * G4 + j] + bi[j] + bh[j];
    const float* w = wh + j * HH;
#pragma unroll 8
    for (int k = 0; k < HH; ++k) acc += hsh[k] * w[k];
    gates[j] = acc;
    __syncthreads();

    if (j < HH) {
        float ig = gates[j], fg = gates[HH + j], gg = gates[2 * HH + j], og = gates[3 * HH + j];
        float c  = sigmf(fg) * cstate[b * HH + j] + sigmf(ig) * tanhf(gg);
        float hn = sigmf(og) * tanhf(c);
        cstate[b * HH + j] = c;
        hstate[b * HH + j] = hn;
        hs_out[(t * BATCH + b) * HH + j] = hn;
    }
}

// ------------------------------ final readout ------------------------------
__global__ void final_linear(const float* __restrict__ h2, const float* __restrict__ wl,
                             const float* __restrict__ bl, float* __restrict__ out) {
    int i = blockIdx.x * blockDim.x + threadIdx.x;     // BATCH*OUTC
    if (i >= BATCH * OUTC) return;
    int b = i / OUTC, m = i - b * OUTC;
    const float* hp = h2 + b * HH;
    const float* w  = wl + m * HH;
    float acc = bl[m];
#pragma unroll 8
    for (int k = 0; k < HH; ++k) acc += hp[k] * w[k];
    out[i] = acc;   // [B, NNODE*2] flat == [B*NNODE, 2] flat
}

// ------------------------------- host driver -------------------------------

static inline void* bump(char*& p, size_t bytes) {
    void* r = (void*)p;
    p += (bytes + 255) & ~(size_t)255;
    return r;
}

extern "C" void kernel_launch(void* const* d_in, const int* in_sizes, int n_in,
                              void* d_out, int out_size, void* d_ws, size_t ws_size,
                              hipStream_t stream) {
    (void)n_in; (void)out_size; (void)ws_size;
    const float* x     = (const float*)d_in[0];
    const int*   ei    = (const int*)  d_in[1];
    const float* w0    = (const float*)d_in[2];
    const float* asrc0 = (const float*)d_in[3];
    const float* adst0 = (const float*)d_in[4];
    const float* b0    = (const float*)d_in[5];
    const float* w1    = (const float*)d_in[6];
    const float* asrc1 = (const float*)d_in[7];
    const float* adst1 = (const float*)d_in[8];
    const float* b1    = (const float*)d_in[9];
    const float* wi1   = (const float*)d_in[10];
    const float* wh1   = (const float*)d_in[11];
    const float* bi1   = (const float*)d_in[12];
    const float* bh1   = (const float*)d_in[13];
    const float* wi2   = (const float*)d_in[14];
    const float* wh2   = (const float*)d_in[15];
    const float* bi2   = (const float*)d_in[16];
    const float* bh2   = (const float*)d_in[17];
    const float* wl    = (const float*)d_in[18];
    const float* bl    = (const float*)d_in[19];
    float* out = (float*)d_out;

    const int E    = in_sizes[1] / 2;
    const int Eall = E + N_NODES;

    char* wp = (char*)d_ws;
    bf16_t*  xb     = (bf16_t*) bump(wp, (size_t)N_NODES * FIN * 2);
    bf16_t*  w0tb   = (bf16_t*) bump(wp, (size_t)FIN * HDIM * 2);   // [HDIM, FIN]
    bf16_t*  w1tb   = (bf16_t*) bump(wp, (size_t)FIN * HDIM * 2);   // [HDIM, FIN]
    bf16_t*  wi1b   = (bf16_t*) bump(wp, (size_t)G4 * SEQ_IN * 2);  // [G4, SEQ_IN] native
    bf16_t*  wi2b   = (bf16_t*) bump(wp, (size_t)G4 * HH * 2);      // [G4, HH] native
    float*   hfull  = (float*)  bump(wp, (size_t)N_NODES * HDIM * 4);
    float*   as_    = (float*)  bump(wp, (size_t)N_NODES * HEADS * 4);
    float*   ad_    = (float*)  bump(wp, (size_t)N_NODES * HEADS * 4);
    unsigned* emaxU = (unsigned*)bump(wp, (size_t)N_NODES * HEADS * 4);
    float*   denom  = (float*)  bump(wp, (size_t)N_NODES * HEADS * 4);
    float*   eBuf   = (float*)  bump(wp, (size_t)Eall * HEADS * 4);
    float*   x1     = (float*)  bump(wp, (size_t)N_NODES * FIN * 4);
    bf16_t*  x1b    = (bf16_t*) bump(wp, (size_t)N_NODES * FIN * 2);
    float*   x2     = (float*)  bump(wp, (size_t)N_NODES * FIN * 4);
    bf16_t*  Xseqb  = (bf16_t*) bump(wp, (size_t)TBROWS * SEQ_IN * 2);
    float*   G1     = (float*)  bump(wp, (size_t)TBROWS * G4 * 4);
    float*   hs1    = (float*)  bump(wp, (size_t)TBROWS * HH * 4);
    bf16_t*  hs1b   = (bf16_t*) bump(wp, (size_t)TBROWS * HH * 2);
    float*   G2     = (float*)  bump(wp, (size_t)TBROWS * G4 * 4);
    float*   hs2    = (float*)  bump(wp, (size_t)TBROWS * HH * 4);
    float*   h1s    = (float*)  bump(wp, (size_t)BATCH * HH * 4);
    float*   c1s    = (float*)  bump(wp, (size_t)BATCH * HH * 4);
    float*   h2s    = (float*)  bump(wp, (size_t)BATCH * HH * 4);
    float*   c2s    = (float*)  bump(wp, (size_t)BATCH * HH * 4);

    const int TPB = 256;
    auto blocks = [](long long n, int t) { return (unsigned)((n + t - 1) / t); };

    // --- operand conversions (once) ---
    cvt_f32_bf16<<<blocks((long long)N_NODES * FIN, TPB), TPB, 0, stream>>>(x, xb, N_NODES * FIN);
    cvt_f32_bf16_T<<<blocks(FIN * HDIM, TPB), TPB, 0, stream>>>(w0, w0tb, FIN, HDIM);
    cvt_f32_bf16_T<<<blocks(FIN * HDIM, TPB), TPB, 0, stream>>>(w1, w1tb, FIN, HDIM);
    cvt_f32_bf16<<<blocks((long long)G4 * SEQ_IN, TPB), TPB, 0, stream>>>(wi1, wi1b, G4 * SEQ_IN);
    cvt_f32_bf16<<<blocks(G4 * HH, TPB), TPB, 0, stream>>>(wi2, wi2b, G4 * HH);

    // --- one GAT layer ---
    auto gat_layer = [&](const bf16_t* Ab, const bf16_t* Wtb, const float* att_s,
                         const float* att_d, const float* bias, float* outx) {
        dim3 g(HDIM / 32, N_NODES / 32);
        wmma_gemm_bf16_nt<<<g, 32, 0, stream>>>(Ab, Wtb, hfull, N_NODES, HDIM, FIN);
        gat_alpha<<<blocks(N_NODES * HEADS, TPB), TPB, 0, stream>>>(hfull, att_s, att_d, as_, ad_);
        fill_u32<<<blocks(N_NODES * HEADS, TPB), TPB, 0, stream>>>(emaxU, 0x007FFFFFu, N_NODES * HEADS);
        fill_f32<<<blocks(N_NODES * HEADS, TPB), TPB, 0, stream>>>(denom, 0.f, N_NODES * HEADS);
        fill_f32<<<blocks((long long)N_NODES * FIN, TPB), TPB, 0, stream>>>(outx, 0.f, N_NODES * FIN);
        gat_edge_logits<<<blocks(Eall, TPB), TPB, 0, stream>>>(ei, E, as_, ad_, eBuf, emaxU);
        gat_edge_exp<<<blocks(Eall, TPB), TPB, 0, stream>>>(ei, E, eBuf, emaxU, denom);
        gat_scatter<<<blocks((long long)Eall * FIN, TPB), TPB, 0, stream>>>(ei, E, eBuf, denom, hfull, outx);
        add_bias_rows<<<blocks((long long)N_NODES * FIN, TPB), TPB, 0, stream>>>(outx, bias, N_NODES * FIN, FIN);
    };

    gat_layer(xb,  w0tb, asrc0, adst0, b0, x1);
    cvt_f32_bf16<<<blocks((long long)N_NODES * FIN, TPB), TPB, 0, stream>>>(x1, x1b, N_NODES * FIN);
    gat_layer(x1b, w1tb, asrc1, adst1, b1, x2);

    // --- sequence formation + batched LSTM1 input projection ---
    reshape_to_seq<<<blocks((long long)TBROWS * SEQ_IN, TPB), TPB, 0, stream>>>(x2, Xseqb);
    {
        dim3 g(G4 / 32, TBROWS / 32);
        wmma_gemm_bf16_nt<<<g, 32, 0, stream>>>(Xseqb, wi1b, G1, TBROWS, G4, SEQ_IN);
    }

    // --- LSTM1 recurrence ---
    fill_f32<<<blocks(BATCH * HH, TPB), TPB, 0, stream>>>(h1s, 0.f, BATCH * HH);
    fill_f32<<<blocks(BATCH * HH, TPB), TPB, 0, stream>>>(c1s, 0.f, BATCH * HH);
    for (int t = 0; t < T_STEPS; ++t)
        lstm_step<<<BATCH, G4, 0, stream>>>(G1, t, wh1, bi1, bh1, h1s, c1s, hs1);

    // --- LSTM2 input projection (batched over T) + recurrence ---
    cvt_f32_bf16<<<blocks(TBROWS * HH, TPB), TPB, 0, stream>>>(hs1, hs1b, TBROWS * HH);
    {
        dim3 g(G4 / 32, TBROWS / 32);
        wmma_gemm_bf16_nt<<<g, 32, 0, stream>>>(hs1b, wi2b, G2, TBROWS, G4, HH);
    }
    fill_f32<<<blocks(BATCH * HH, TPB), TPB, 0, stream>>>(h2s, 0.f, BATCH * HH);
    fill_f32<<<blocks(BATCH * HH, TPB), TPB, 0, stream>>>(c2s, 0.f, BATCH * HH);
    for (int t = 0; t < T_STEPS; ++t)
        lstm_step<<<BATCH, G4, 0, stream>>>(G2, t, wh2, bi2, bh2, h2s, c2s, hs2);

    // --- readout ---
    final_linear<<<blocks(BATCH * OUTC, TPB), TPB, 0, stream>>>(h2s, wl, bl, out);
}
